// MultiheadAttention_52269751992979
// MI455X (gfx1250) — matrix operand order repaired and checked by reading, not compiled, not run on hardware
//
#include <hip/hip_runtime.h>
#include <hip/hip_bf16.h>

// ---------------------------------------------------------------------------
// MultiheadAttention forward for MI455X (gfx1250), wave32 + WMMA bf16.
//   qkv = x @ Win^T + bin ; q *= D^-0.5
//   flash attention per (b,h) with online softmax
//   out = attn @ Wout^T + bout
// GEMMs: v_wmma_f32_16x16x32_bf16 (f32 accumulate), BK=64 per barrier epoch.
// Tile staging: GLOBAL_LOAD_ASYNC_TO_LDS_B128 (ASYNCcnt) when available.
// ---------------------------------------------------------------------------

typedef __attribute__((ext_vector_type(16))) __bf16 v16bf;
typedef __attribute__((ext_vector_type(8)))  __bf16 v8bf;
typedef __attribute__((ext_vector_type(4)))  __bf16 v4bf;
typedef __attribute__((ext_vector_type(8)))  float  v8f;
typedef __attribute__((ext_vector_type(4)))  int    v4i;

#define L_SEQ  2048
#define BATCH  4
#define EMB    1024
#define HEADS  16
#define DHEAD  64
#define M_ROWS (L_SEQ * BATCH)   // 8192
#define N_QKV  (3 * EMB)         // 3072
#define SCALING 0.125f           // 64^-0.5
#define TSTRIDE 72               // 64 + 8 bf16 pad; row stride 144B (16B aligned)

#ifndef __has_builtin
#define __has_builtin(x) 0
#endif
#if __has_builtin(__builtin_amdgcn_global_load_async_to_lds_b128)
#define HAVE_ASYNC_LDS 1
#else
#define HAVE_ASYNC_LDS 0
#endif

#if HAVE_ASYNC_LDS
typedef __attribute__((address_space(1))) v4i as1_v4i;   // global
typedef __attribute__((address_space(3))) v4i as3_v4i;   // LDS
#endif

// 16B global -> LDS copy. Async (no VGPR data, ASYNCcnt) on gfx1250.
static __device__ __forceinline__ void async_copy16(const __bf16* gsrc, __bf16* ldst) {
#if HAVE_ASYNC_LDS
  __builtin_amdgcn_global_load_async_to_lds_b128(
      (as1_v4i*)(uintptr_t)gsrc,
      (as3_v4i*)(unsigned)(uintptr_t)ldst, 0, 0);
#else
  *(v8bf*)ldst = *(const v8bf*)gsrc;
#endif
}

static __device__ __forceinline__ void wait_async_copies() {
#if HAVE_ASYNC_LDS
#if __has_builtin(__builtin_amdgcn_s_wait_asynccnt)
  __builtin_amdgcn_s_wait_asynccnt(0);
#else
  asm volatile("s_wait_asynccnt 0x0" ::: "memory");
#endif
#endif
}

// round-to-nearest-even f32 -> bf16
static __device__ __forceinline__ __bf16 f2bf(float f) {
  unsigned u = __builtin_bit_cast(unsigned, f);
  unsigned r = u + 0x7FFFu + ((u >> 16) & 1u);
  unsigned short h = (unsigned short)(r >> 16);
  return __builtin_bit_cast(__bf16, h);
}

// A-matrix 16x32 bf16 fragment (ISA 7.12.2): lane m=l&15; lanes<16 k0=0, lanes>=16 k0=8.
// frag elems 0..7 = A[m, kb..kb+7], elems 8..15 = A[m, kb+16..kb+23]  (two 16B runs).
static __device__ __forceinline__ v16bf ld_frag_a(const __bf16* base, int stride,
                                                  int row0, int koff, int lane) {
  int r  = lane & 15;
  int kb = koff + ((lane & 16) ? 8 : 0);
  const __bf16* p = base + (row0 + r) * stride + kb;
  v8bf lo = *(const v8bf*)(p);
  v8bf hi = *(const v8bf*)(p + 16);
  return __builtin_shufflevector(lo, hi, 0,1,2,3,4,5,6,7,8,9,10,11,12,13,14,15);
}

// B-matrix 32x16 bf16 fragment: lane n=l&15; lanes<16 k0=0, lanes>=16 k0=8.
// frag elems 0..7 = B[kb..kb+7, n], elems 8..15 = B[kb+16..kb+23, n]; LDS tile stored "column-major" Bt[n][k].
static __device__ __forceinline__ v16bf ld_frag_b(const __bf16* base, int stride,
                                                  int col0, int koff, int lane) {
  int n  = lane & 15;
  int kb = koff + ((lane & 16) ? 8 : 0);
  const __bf16* p = base + (col0 + n) * stride + kb;
  v8bf lo = *(const v8bf*)(p);
  v8bf hi = *(const v8bf*)(p + 16);
  return __builtin_shufflevector(lo, hi, 0,1,2,3,4,5,6,7,8,9,10,11,12,13,14,15);
}

static __device__ __forceinline__ v8f wmma_bf16(v16bf a, v16bf b, v8f c) {
  return __builtin_amdgcn_wmma_f32_16x16x32_bf16(false, a, false, b,
                                                 (short)0, c, false, false);
}

// ---------------------------------------------------------------------------
__global__ void cast_f32_to_bf16(const float* __restrict__ src,
                                 __bf16* __restrict__ dst, int n4) {
  int i = blockIdx.x * blockDim.x + threadIdx.x;
  if (i >= n4) return;
  float4 v = ((const float4*)src)[i];
  v4bf o;
  o.x = f2bf(v.x); o.y = f2bf(v.y); o.z = f2bf(v.z); o.w = f2bf(v.w);
  ((v4bf*)dst)[i] = o;
}

// ---------------------------------------------------------------------------
// QKV GEMM: C[8192,3072] = X[8192,1024] * Win^T, bias + scaling fused,
// scattered to Q[B,H,L,D], K[B,H,L,D], Vt[B,H,D,L] in bf16.
// Block tile 128(M) x 64(N) x BK=64; 8 waves each computing 32x32.
__global__ __launch_bounds__(256) void qkv_gemm(
    const __bf16* __restrict__ X, const __bf16* __restrict__ W,
    const float* __restrict__ bias,
    __bf16* __restrict__ Qt, __bf16* __restrict__ Kt, __bf16* __restrict__ Vt) {
  __shared__ __align__(16) __bf16 As[128 * TSTRIDE];
  __shared__ __align__(16) __bf16 Bs[64 * TSTRIDE];

  const int tid  = threadIdx.x;
  const int lane = tid & 31;
  const int wid  = tid >> 5;
  const int m0 = blockIdx.y * 128;
  const int n0 = blockIdx.x * 64;
  const int mb = (wid & 3) * 32;
  const int nb = (wid >> 2) * 32;

  v8f acc[2][2];
  const v8f vz = {0.f,0.f,0.f,0.f,0.f,0.f,0.f,0.f};
  acc[0][0] = vz; acc[0][1] = vz; acc[1][0] = vz; acc[1][1] = vz;

  for (int k0 = 0; k0 < EMB; k0 += 64) {
#pragma unroll
    for (int i = 0; i < 4; ++i) {   // A tile: 128 rows x 64 k, 1024 16B chunks
      int c = tid + 256 * i;
      int row = c >> 3, off = (c & 7) * 8;
      async_copy16(X + (size_t)(m0 + row) * EMB + k0 + off, &As[row * TSTRIDE + off]);
    }
#pragma unroll
    for (int i = 0; i < 2; ++i) {   // B tile: 64 "cols" x 64 k (Win rows)
      int c = tid + 256 * i;
      int row = c >> 3, off = (c & 7) * 8;
      async_copy16(W + (size_t)(n0 + row) * EMB + k0 + off, &Bs[row * TSTRIDE + off]);
    }
    if (k0 + 64 < EMB) {            // prefetch next A slice (global_prefetch_b8)
      int row = tid >> 1;
      __builtin_prefetch(X + (size_t)(m0 + row) * EMB + k0 + 64, 0, 1);
    }
    wait_async_copies();
    __syncthreads();

#pragma unroll
    for (int kk = 0; kk < 64; kk += 32) {
      v16bf a0 = ld_frag_a(As, TSTRIDE, mb,      kk, lane);
      v16bf a1 = ld_frag_a(As, TSTRIDE, mb + 16, kk, lane);
      v16bf b0 = ld_frag_b(Bs, TSTRIDE, nb,      kk, lane);
      v16bf b1 = ld_frag_b(Bs, TSTRIDE, nb + 16, kk, lane);
      acc[0][0] = wmma_bf16(a0, b0, acc[0][0]);
      acc[0][1] = wmma_bf16(a0, b1, acc[0][1]);
      acc[1][0] = wmma_bf16(a1, b0, acc[1][0]);
      acc[1][1] = wmma_bf16(a1, b1, acc[1][1]);
    }
    __syncthreads();
  }

  // Epilogue: bias, scaling on q, scatter to Q / K / V^T
  const int rsel = (lane & 16) ? 8 : 0;
  const int ncol = lane & 15;
#pragma unroll
  for (int mi = 0; mi < 2; ++mi)
#pragma unroll
    for (int ni = 0; ni < 2; ++ni)
#pragma unroll
      for (int r = 0; r < 8; ++r) {
        int grow = m0 + mb + 16 * mi + r + rsel;        // = l*4 + b
        int gcol = n0 + nb + 16 * ni + ncol;            // 0..3071
        float v = acc[mi][ni][r] + bias[gcol];
        int sect = gcol >> 10;
        int e = gcol & 1023;
        int h = e >> 6, d = e & 63;
        int l = grow >> 2, b = grow & 3;
        size_t bh = (size_t)(b * HEADS + h);
        if (sect == 0) {
          Qt[(bh * L_SEQ + l) * DHEAD + d] = f2bf(v * SCALING);
        } else if (sect == 1) {
          Kt[(bh * L_SEQ + l) * DHEAD + d] = f2bf(v);
        } else {
          Vt[(bh * DHEAD + d) * L_SEQ + l] = f2bf(v);   // transposed
        }
      }
}

// ---------------------------------------------------------------------------
// Flash attention: one block = 64 query rows of one (b,h); 4 waves, each owning
// 16 query rows. Loops over 32 key blocks of 64. Online softmax in LDS.
__global__ __launch_bounds__(128) void flash_attn(
    const __bf16* __restrict__ Qt, const __bf16* __restrict__ Kt,
    const __bf16* __restrict__ Vt, const unsigned char* __restrict__ kpm,
    __bf16* __restrict__ X2) {
  __shared__ __align__(16) __bf16 Qs [64 * TSTRIDE];
  __shared__ __align__(16) __bf16 Ks [64 * TSTRIDE];
  __shared__ __align__(16) __bf16 Vts[64 * TSTRIDE];
  __shared__ __align__(16) __bf16 Ps [64 * TSTRIDE];
  __shared__ float Ss[64 * 65];
  __shared__ float mrow[64], lrow[64], crow[64];

  const int tid  = threadIdx.x;
  const int lane = tid & 31;
  const int wid  = tid >> 5;
  const int bh = blockIdx.y;           // 0..63
  const int b  = bh >> 4, h = bh & 15;
  const int q0 = blockIdx.x * 64;

  const __bf16* qbase  = Qt + (size_t)bh * L_SEQ * DHEAD;
  const __bf16* kbase  = Kt + (size_t)bh * L_SEQ * DHEAD;
  const __bf16* vtbase = Vt + (size_t)bh * DHEAD * L_SEQ;

#pragma unroll
  for (int i = 0; i < 4; ++i) {        // load Q block once: 64x64 bf16
    int c = tid + 128 * i;             // 512 chunks of 16B
    int row = c >> 3, off = (c & 7) * 8;
    async_copy16(qbase + (size_t)(q0 + row) * DHEAD + off, &Qs[row * TSTRIDE + off]);
  }
  if (tid < 64) { mrow[tid] = -1e30f; lrow[tid] = 0.f; }

  v8f o[4];
  const v8f vz = {0.f,0.f,0.f,0.f,0.f,0.f,0.f,0.f};
  o[0] = vz; o[1] = vz; o[2] = vz; o[3] = vz;

  const int rsel = (lane & 16) ? 8 : 0;
  const int ncol = lane & 15;

  for (int t = 0; t < L_SEQ / 64; ++t) {
    const int s0 = t * 64;
#pragma unroll
    for (int i = 0; i < 4; ++i) {      // load K block (rows s, contiguous d)
      int c = tid + 128 * i;
      int row = c >> 3, off = (c & 7) * 8;
      async_copy16(kbase + (size_t)(s0 + row) * DHEAD + off, &Ks[row * TSTRIDE + off]);
    }
#pragma unroll
    for (int i = 0; i < 4; ++i) {      // load V^T block (rows d, contiguous s)
      int c = tid + 128 * i;
      int row = c >> 3, off = (c & 7) * 8;
      async_copy16(vtbase + (size_t)row * L_SEQ + s0 + off, &Vts[row * TSTRIDE + off]);
    }
    wait_async_copies();
    __syncthreads();

    // S(16 rows x 64 cols) = Q * K^T  (Q pre-scaled)
    v8f s4[4];
    s4[0] = vz; s4[1] = vz; s4[2] = vz; s4[3] = vz;
#pragma unroll
    for (int kk = 0; kk < 64; kk += 32) {
      v16bf a = ld_frag_a(Qs, TSTRIDE, 16 * wid, kk, lane);
#pragma unroll
      for (int nt = 0; nt < 4; ++nt)
        s4[nt] = wmma_bf16(a, ld_frag_b(Ks, TSTRIDE, 16 * nt, kk, lane), s4[nt]);
    }
#pragma unroll
    for (int nt = 0; nt < 4; ++nt)
#pragma unroll
      for (int r = 0; r < 8; ++r)
        Ss[(16 * wid + r + rsel) * 65 + 16 * nt + ncol] = s4[nt][r];
    __syncthreads();

    // Online softmax: one thread per query row
    if (tid < 64) {
      const int q = tid;
      float mo = mrow[q];
      float mx = -1e30f;
      for (int s = 0; s < 64; ++s) {
        float v = Ss[q * 65 + s];
        if (kpm[(size_t)(s0 + s) * BATCH + b]) v = -1e30f;
        Ss[q * 65 + s] = v;
        mx = fmaxf(mx, v);
      }
      float mn = fmaxf(mo, mx);
      float corr = __expf(mo - mn);
      float sum = 0.f;
      for (int s = 0; s < 64; ++s) {
        float p = __expf(Ss[q * 65 + s] - mn);
        Ps[q * TSTRIDE + s] = f2bf(p);
        sum += p;
      }
      lrow[q] = lrow[q] * corr + sum;
      mrow[q] = mn;
      crow[q] = corr;
    }
    __syncthreads();

    // rescale O and accumulate O += P * V   (V^T rows = d, k = s)
#pragma unroll
    for (int r = 0; r < 8; ++r) {
      float cr = crow[16 * wid + r + rsel];
#pragma unroll
      for (int dt = 0; dt < 4; ++dt) o[dt][r] *= cr;
    }
#pragma unroll
    for (int kk = 0; kk < 64; kk += 32) {
      v16bf a = ld_frag_a(Ps, TSTRIDE, 16 * wid, kk, lane);
#pragma unroll
      for (int dt = 0; dt < 4; ++dt)
        o[dt] = wmma_bf16(a, ld_frag_b(Vts, TSTRIDE, 16 * dt, kk, lane), o[dt]);
    }
    __syncthreads();
  }

  // Final: O / l, write bf16 attn matrix X2[m = l*4+b, e = h*64+d]
#pragma unroll
  for (int r = 0; r < 8; ++r) {
    int qrow = 16 * wid + r + rsel;
    float inv = 1.0f / lrow[qrow];
    int q = q0 + qrow;
#pragma unroll
    for (int dt = 0; dt < 4; ++dt) {
      int d = 16 * dt + ncol;
      X2[((size_t)q * BATCH + b) * EMB + h * DHEAD + d] = f2bf(o[dt][r] * inv);
    }
  }
}

// ---------------------------------------------------------------------------
// Out projection: out[8192,1024] = X2 * Wout^T + bout   (f32 output)
__global__ __launch_bounds__(256) void out_gemm(
    const __bf16* __restrict__ X, const __bf16* __restrict__ W,
    const float* __restrict__ bias, float* __restrict__ out) {
  __shared__ __align__(16) __bf16 As[128 * TSTRIDE];
  __shared__ __align__(16) __bf16 Bs[64 * TSTRIDE];

  const int tid  = threadIdx.x;
  const int lane = tid & 31;
  const int wid  = tid >> 5;
  const int m0 = blockIdx.y * 128;
  const int n0 = blockIdx.x * 64;
  const int mb = (wid & 3) * 32;
  const int nb = (wid >> 2) * 32;

  v8f acc[2][2];
  const v8f vz = {0.f,0.f,0.f,0.f,0.f,0.f,0.f,0.f};
  acc[0][0] = vz; acc[0][1] = vz; acc[1][0] = vz; acc[1][1] = vz;

  for (int k0 = 0; k0 < EMB; k0 += 64) {
#pragma unroll
    for (int i = 0; i < 4; ++i) {
      int c = tid + 256 * i;
      int row = c >> 3, off = (c & 7) * 8;
      async_copy16(X + (size_t)(m0 + row) * EMB + k0 + off, &As[row * TSTRIDE + off]);
    }
#pragma unroll
    for (int i = 0; i < 2; ++i) {
      int c = tid + 256 * i;
      int row = c >> 3, off = (c & 7) * 8;
      async_copy16(W + (size_t)(n0 + row) * EMB + k0 + off, &Bs[row * TSTRIDE + off]);
    }
    if (k0 + 64 < EMB) {
      int row = tid >> 1;
      __builtin_prefetch(X + (size_t)(m0 + row) * EMB + k0 + 64, 0, 1);
    }
    wait_async_copies();
    __syncthreads();

#pragma unroll
    for (int kk = 0; kk < 64; kk += 32) {
      v16bf a0 = ld_frag_a(As, TSTRIDE, mb,      kk, lane);
      v16bf a1 = ld_frag_a(As, TSTRIDE, mb + 16, kk, lane);
      v16bf b0 = ld_frag_b(Bs, TSTRIDE, nb,      kk, lane);
      v16bf b1 = ld_frag_b(Bs, TSTRIDE, nb + 16, kk, lane);
      acc[0][0] = wmma_bf16(a0, b0, acc[0][0]);
      acc[0][1] = wmma_bf16(a0, b1, acc[0][1]);
      acc[1][0] = wmma_bf16(a1, b0, acc[1][0]);
      acc[1][1] = wmma_bf16(a1, b1, acc[1][1]);
    }
    __syncthreads();
  }

  const int rsel = (lane & 16) ? 8 : 0;
  const int ncol = lane & 15;
#pragma unroll
  for (int mi = 0; mi < 2; ++mi)
#pragma unroll
    for (int ni = 0; ni < 2; ++ni)
#pragma unroll
      for (int r = 0; r < 8; ++r) {
        int grow = m0 + mb + 16 * mi + r + rsel;
        int gcol = n0 + nb + 16 * ni + ncol;
        out[(size_t)grow * EMB + gcol] = acc[mi][ni][r] + bias[gcol];
      }
}

// ---------------------------------------------------------------------------
extern "C" void kernel_launch(void* const* d_in, const int* in_sizes, int n_in,
                              void* d_out, int out_size, void* d_ws, size_t ws_size,
                              hipStream_t stream) {
  const float* x     = (const float*)d_in[0];           // [L,B,E]
  const float* w_in  = (const float*)d_in[1];           // [3E,E]
  const float* b_in  = (const float*)d_in[2];           // [3E]
  const float* w_out = (const float*)d_in[3];           // [E,E]
  const float* b_out = (const float*)d_in[4];           // [E]
  const unsigned char* kpm = (const unsigned char*)d_in[5]; // [L,B] bool
  float* out = (float*)d_out;

  char* ws = (char*)d_ws;
  const size_t SZ_XBF  = (size_t)M_ROWS * EMB * 2;      // 16 MiB
  const size_t SZ_WIN  = (size_t)N_QKV * EMB * 2;       //  6 MiB
  const size_t SZ_WOUT = (size_t)EMB * EMB * 2;         //  2 MiB
  const size_t SZ_HEAD = (size_t)BATCH * HEADS * L_SEQ * DHEAD * 2; // 16 MiB

  __bf16* Xbf = (__bf16*)(ws);
  __bf16* Wq  = (__bf16*)(ws + SZ_XBF);
  __bf16* Wo  = (__bf16*)(ws + SZ_XBF + SZ_WIN);
  __bf16* Qt  = (__bf16*)(ws + SZ_XBF + SZ_WIN + SZ_WOUT);
  __bf16* Kt  = (__bf16*)(ws + SZ_XBF + SZ_WIN + SZ_WOUT + SZ_HEAD);
  __bf16* Vt  = (__bf16*)(ws + SZ_XBF + SZ_WIN + SZ_WOUT + 2 * SZ_HEAD);
  __bf16* X2  = (__bf16*)(ws + SZ_XBF + SZ_WIN + SZ_WOUT + 3 * SZ_HEAD);

  {
    int n4 = (int)(SZ_XBF / 8);
    cast_f32_to_bf16<<<(n4 + 255) / 256, 256, 0, stream>>>(x, Xbf, n4);
  }
  {
    int n4 = (int)(SZ_WIN / 8);
    cast_f32_to_bf16<<<(n4 + 255) / 256, 256, 0, stream>>>(w_in, Wq, n4);
  }
  {
    int n4 = (int)(SZ_WOUT / 8);
    cast_f32_to_bf16<<<(n4 + 255) / 256, 256, 0, stream>>>(w_out, Wo, n4);
  }

  qkv_gemm<<<dim3(N_QKV / 64, M_ROWS / 128), 256, 0, stream>>>(
      Xbf, Wq, b_in, Qt, Kt, Vt);

  flash_attn<<<dim3(L_SEQ / 64, BATCH * HEADS), 128, 0, stream>>>(
      Qt, Kt, Vt, kpm, X2);

  out_gemm<<<dim3(EMB / 64, M_ROWS / 128), 256, 0, stream>>>(
      X2, Wo, b_out, out);

  (void)in_sizes; (void)n_in; (void)out_size; (void)ws_size;
}
